// SCNN2D_19610820673807
// MI455X (gfx1250) — matrix operand order repaired and checked by reading, not compiled, not run on hardware
//
#include <hip/hip_runtime.h>
#include <hip/hip_bf16.h>

typedef __attribute__((ext_vector_type(16))) _Float16 v16h;
typedef __attribute__((ext_vector_type(8)))  _Float16 v8h;
typedef __attribute__((ext_vector_type(4)))  _Float16 v4h;
typedef __attribute__((ext_vector_type(8)))  float    v8f;

#define CDIV(a,b) (((a)+(b)-1)/(b))

// ======================================================================
// Fused 3x3 conv (pad 1) + bias + BN + LeakyReLU.
// Implicit GEMM: A = weights [Cout x K], B = im2col patches [K x N],
// K = Cin*9 (zero-padded to mult of 32), N = spatial positions of one image.
// Workgroup: 256 threads (8 wave32), tile 32(Cout) x (64*NSUB)(N), K-step 32.
// Each wave owns one A fragment + NSUB B fragments -> NSUB WMMAs / K-step
// (A reused NSUB times; all fragment ds_loads issued before the WMMA group
// so only one dscnt wait gates the whole WMMA burst).
// LDS tiles are double-buffered; global loads for step k+1 are issued
// before the fragment reads / WMMAs of step k (software pipeline).
// All staging loads are branchless (clamped address + select-0).
// NSUB = 4 for layers with H*W % 256 == 0, NSUB = 2 for de4 (H*W = 128).
// ======================================================================
#define BM 32
#define BK 32

template <int NSUB>
__global__ __launch_bounds__(256)
void conv3x3_bn_lrelu(const float* __restrict__ in1, int C1,
                      const float* __restrict__ in2, int C2,
                      const float* __restrict__ Wgt,
                      const float* __restrict__ bias,
                      const float* __restrict__ bn_s, const float* __restrict__ bn_o,
                      const float* __restrict__ bn_m, const float* __restrict__ bn_v,
                      float* __restrict__ out,
                      int Cout, int H, int Wd)
{
  constexpr int BNt   = 64 * NSUB;        // N tile width
  constexpr int TPN   = 256 / BNt;        // threads per n-column (2 or 1)
  constexpr int BELEM = BK / TPN;         // k-elements staged per thread (16 or 32)

  const int Cin    = C1 + C2;
  const int K      = Cin * 9;
  const int Ksteps = (K + BK - 1) / BK;
  const int batch  = blockIdx.z;
  const int m0     = blockIdx.y * BM;
  const int n0     = blockIdx.x * BNt;

  __shared__ __align__(16) _Float16 aT[2][BM][BK + 8];    // weights  [m][k]
  __shared__ __align__(16) _Float16 bT[2][BNt][BK + 8];   // patches  [n][k]
  __shared__ float sAlpha[BM], sBeta[BM];

  const int tid  = threadIdx.x;
  const int lane = tid & 31;
  const int wv   = tid >> 5;          // 8 waves
  const int mt   = wv >> 2;           // 0..1  (16-row slab of Cout tile)
  const int ntb  = wv & 3;            // wave covers N-subtiles ntb + 4*j, j<NSUB
  const int lh   = lane & 15;
  const int hi   = lane >> 4;         // half-wave selector (ISA A/B layout)

  // fold bias+BN into per-channel alpha/beta once per workgroup
  if (tid < BM) {
    int co = m0 + tid;
    float alpha = bn_s[co] * rsqrtf(bn_v[co] + 1e-5f);
    sAlpha[tid] = alpha;
    sBeta[tid]  = (bias[co] - bn_m[co]) * alpha + bn_o[co];
  }

  // A staging: thread -> (m = tid/8, k = (tid%8)*4 .. +3)
  const int am = tid >> 3;
  const int ak = (tid & 7) * 4;
  const float* wrow = Wgt + (size_t)(m0 + am) * K;
  // B staging: thread -> one n column (NSUB=4) or half a column (NSUB=2)
  const int bn_  = tid / TPN;
  const int bk   = (tid % TPN) * BELEM;
  const int p    = n0 + bn_;
  const int ph   = p / Wd;
  const int pw   = p - ph * Wd;

  // ---- branchless staging loads into registers ----
  float areg[4];
  float breg[BELEM];
  auto load_step = [&](int k0) {
    #pragma unroll
    for (int i = 0; i < 4; ++i) {
      int k  = k0 + ak + i;
      int kc = min(k, K - 1);
      float t = wrow[kc];
      areg[i] = (k < K) ? t : 0.0f;
    }
    #pragma unroll
    for (int i = 0; i < BELEM; ++i) {
      int k  = k0 + bk + i;
      int kc = min(k, K - 1);
      int c  = kc / 9;
      int r  = kc - c * 9;
      int r3 = r / 3;
      int y  = ph + r3 - 1;
      int x  = pw + (r - r3 * 3) - 1;
      bool inb = (k < K) & ((unsigned)y < (unsigned)H) & ((unsigned)x < (unsigned)Wd);
      int yc = min(max(y, 0), H - 1);
      int xc = min(max(x, 0), Wd - 1);
      const float* src = (c < C1)
          ? (in1 + (((size_t)batch * C1 + c       ) * H + yc) * Wd + xc)
          : (in2 + (((size_t)batch * C2 + (c - C1)) * H + yc) * Wd + xc);
      float t = *src;
      breg[i] = inb ? t : 0.0f;
    }
  };
  auto store_step = [&](int buf) {
    v4h av;
    #pragma unroll
    for (int i = 0; i < 4; ++i) av[i] = (_Float16)areg[i];
    *(v4h*)&aT[buf][am][ak] = av;                 // ds_store_b64
    #pragma unroll
    for (int v8i = 0; v8i < BELEM / 8; ++v8i) {
      v8h bv;
      #pragma unroll
      for (int i = 0; i < 8; ++i) bv[i] = (_Float16)breg[v8i * 8 + i];
      *(v8h*)&bT[buf][bn_][bk + v8i * 8] = bv;    // ds_store_b128
    }
  };

  v8f acc[NSUB];
  #pragma unroll
  for (int j = 0; j < NSUB; ++j) acc[j] = (v8f){};

  // prologue: stage K-step 0 into buffer 0
  load_step(0);
  store_step(0);
  __syncthreads();

  for (int ks = 0; ks < Ksteps; ++ks) {
    const int cur = ks & 1;
    const int nxt = cur ^ 1;
    const bool more = (ks + 1 < Ksteps);
    if (more) {
      __builtin_prefetch(wrow + (ks + 2) * BK + ak, 0, 1);  // global_prefetch_b8
      load_step((ks + 1) * BK);                             // overlap with WMMA below
    }

    // ---- load ALL fragments first (one ds_load burst), then WMMA burst ----
    // A 16x32: lanes 0-15 hold M=lh, K {0..7,16..23}; lanes 16-31 K {8..15,24..31}
    const _Float16* arow = &aT[cur][mt * 16 + lh][0];
    v8h a0 = *(const v8h*)(arow + hi * 8);
    v8h a1 = *(const v8h*)(arow + 16 + hi * 8);
    v16h afrag = __builtin_shufflevector(a0, a1, 0,1,2,3,4,5,6,7,8,9,10,11,12,13,14,15);
    // B 32x16: lanes 0-15 hold N=lh, K {0..7,16..23}; lanes 16-31 K {8..15,24..31}
    v16h bfrag[NSUB];
    #pragma unroll
    for (int j = 0; j < NSUB; ++j) {
      const _Float16* brow = &bT[cur][(ntb + 4 * j) * 16 + lh][0];
      v8h b0 = *(const v8h*)(brow + hi * 8);
      v8h b1 = *(const v8h*)(brow + 16 + hi * 8);
      bfrag[j] = __builtin_shufflevector(b0, b1, 0,1,2,3,4,5,6,7,8,9,10,11,12,13,14,15);
    }
    #pragma unroll
    for (int j = 0; j < NSUB; ++j) {
      acc[j] = __builtin_amdgcn_wmma_f32_16x16x32_f16(false, afrag, false, bfrag[j],
                                                      (short)0, acc[j], false, false);
    }

    if (more) store_step(nxt);
    __syncthreads();
  }

  // ---- epilogue: D layout lanes0-15 M=r, lanes16-31 M=r+8; N = lane&15 ----
  #pragma unroll
  for (int j = 0; j < NSUB; ++j) {
    const int n = n0 + (ntb + 4 * j) * 16 + lh;
    const int h = n / Wd;
    const int w = n - h * Wd;
    #pragma unroll
    for (int r = 0; r < 8; ++r) {
      int m   = mt * 16 + hi * 8 + r;
      float y = acc[j][r] * sAlpha[m] + sBeta[m];
      y = (y > 0.0f) ? y : 0.01f * y;
      out[(((size_t)batch * Cout + (m0 + m)) * H + h) * Wd + w] = y;
    }
  }
}

// ======================================================================
// 2x2 max pool
// ======================================================================
__global__ void maxpool2_kernel(const float* __restrict__ in, float* __restrict__ out,
                                int H, int W, int total)
{
  int i = blockIdx.x * blockDim.x + threadIdx.x;
  if (i >= total) return;
  int Wh = W >> 1, Hh = H >> 1;
  int w  = i % Wh;
  int t  = i / Wh;
  int h  = t % Hh;
  int bc = t / Hh;
  const float* ptr = in + ((size_t)bc * H + 2 * h) * W + 2 * w;
  out[i] = fmaxf(fmaxf(ptr[0], ptr[1]), fmaxf(ptr[W], ptr[W + 1]));
}

// ======================================================================
// 2x bilinear upsample with the reference's index mapping
// ======================================================================
__global__ void up2x_kernel(const float* __restrict__ in, float* __restrict__ out,
                            int H, int W, int total)
{
  int i = blockIdx.x * blockDim.x + threadIdx.x;
  if (i >= total) return;
  int W2 = 2 * W, H2 = 2 * H;
  int ox = i % W2;
  int t  = i / W2;
  int oy = t % H2;
  int bc = t / H2;
  float cy = (float)oy * ((float)(H - 1) / (float)(2 * H - 1));
  float cx = (float)ox * ((float)(W - 1) / (float)(2 * W - 1));
  int y0 = (int)floorf(cy);  int y1 = min(y0 + 1, H - 1);  float fy = cy - (float)y0;
  int x0 = (int)floorf(cx);  int x1 = min(x0 + 1, W - 1);  float fx = cx - (float)x0;
  const float* ptr = in + (size_t)bc * H * W;
  float v00 = ptr[y0 * W + x0], v01 = ptr[y0 * W + x1];
  float v10 = ptr[y1 * W + x0], v11 = ptr[y1 * W + x1];
  float r0 = v00 * (1.0f - fy) + v10 * fy;
  float r1 = v01 * (1.0f - fy) + v11 * fy;
  out[i] = r0 * (1.0f - fx) + r1 * fx;
}

// ======================================================================
// SCNN row scan (recurrence over H, 5-tap conv over W). C = 32 fixed.
// One workgroup per batch image; recurrence row + weights live in LDS.
// dir > 0: rows 1..H-1;  dir < 0: rows H-2..1 (ends kept).
// ======================================================================
__global__ __launch_bounds__(256)
void scan_rows_kernel(float* __restrict__ top, const float* __restrict__ wgt,
                      int H, int W, int dir)
{
  const int b   = blockIdx.x;
  const int tid = threadIdx.x;
  __shared__ float prev[32 * 256];    // C*W  (W <= 256)
  __shared__ float wl[32 * 32 * 5];

  for (int i = tid; i < 32 * 32 * 5; i += 256) wl[i] = wgt[i];
  const int h0 = (dir > 0) ? 0 : H - 1;
  for (int i = tid; i < 32 * W; i += 256) {
    int c = i / W, w = i - c * W;
    prev[c * W + w] = top[(((size_t)b * 32 + c) * H + h0) * W + w];
  }
  __syncthreads();

  const int nsteps = (dir > 0) ? (H - 1) : (H - 2);
  float nv[32];
  for (int s = 0; s < nsteps; ++s) {
    int h = (dir > 0) ? (1 + s) : (H - 2 - s);
    int j = 0;
    for (int e = tid; e < 32 * W; e += 256, ++j) {
      int c = e / W, w = e - c * W;
      float a = 0.0f;
      const float* wc = &wl[c * 160];
      #pragma unroll 4
      for (int ci = 0; ci < 32; ++ci) {
        const float* pr = &prev[ci * W];
        const float* wk = &wc[ci * 5];
        #pragma unroll
        for (int k = 0; k < 5; ++k) {
          int x = w + k - 2;
          if ((unsigned)x < (unsigned)W) a += wk[k] * pr[x];
        }
      }
      a = fmaxf(a, 0.0f);
      nv[j] = top[(((size_t)b * 32 + c) * H + h) * W + w] + a;
    }
    __syncthreads();
    j = 0;
    for (int e = tid; e < 32 * W; e += 256, ++j) {
      int c = e / W, w = e - c * W;
      prev[c * W + w] = nv[j];
      top[(((size_t)b * 32 + c) * H + h) * W + w] = nv[j];
    }
    __syncthreads();
  }
}

// ======================================================================
// SCNN column scan (recurrence over W, 5-tap conv over H). C = 32 fixed.
// ======================================================================
__global__ __launch_bounds__(256)
void scan_cols_kernel(float* __restrict__ top, const float* __restrict__ wgt,
                      int H, int W, int dir)
{
  const int b   = blockIdx.x;
  const int tid = threadIdx.x;
  __shared__ float prev[32 * 128];    // C*H  (H <= 128)
  __shared__ float wl[32 * 32 * 5];

  for (int i = tid; i < 32 * 32 * 5; i += 256) wl[i] = wgt[i];
  const int w0 = (dir > 0) ? 0 : W - 1;
  for (int i = tid; i < 32 * H; i += 256) {
    int c = i / H, h = i - c * H;
    prev[c * H + h] = top[(((size_t)b * 32 + c) * H + h) * W + w0];
  }
  __syncthreads();

  const int nsteps = (dir > 0) ? (W - 1) : (W - 2);
  float nv[32];
  for (int s = 0; s < nsteps; ++s) {
    int w = (dir > 0) ? (1 + s) : (W - 2 - s);
    int j = 0;
    for (int e = tid; e < 32 * H; e += 256, ++j) {
      int c = e / H, h = e - c * H;
      float a = 0.0f;
      const float* wc = &wl[c * 160];
      #pragma unroll 4
      for (int ci = 0; ci < 32; ++ci) {
        const float* pr = &prev[ci * H];
        const float* wk = &wc[ci * 5];
        #pragma unroll
        for (int k = 0; k < 5; ++k) {
          int y = h + k - 2;
          if ((unsigned)y < (unsigned)H) a += wk[k] * pr[y];
        }
      }
      a = fmaxf(a, 0.0f);
      nv[j] = top[(((size_t)b * 32 + c) * H + h) * W + w] + a;
    }
    __syncthreads();
    j = 0;
    for (int e = tid; e < 32 * H; e += 256, ++j) {
      int c = e / H, h = e - c * H;
      prev[c * H + h] = nv[j];
      top[(((size_t)b * 32 + c) * H + h) * W + w] = nv[j];
    }
    __syncthreads();
  }
}

// ======================================================================
// 1x1 output conv (32 -> 1) + sigmoid
// ======================================================================
__global__ void outconv_sigmoid_kernel(const float* __restrict__ top,
                                       const float* __restrict__ w,
                                       const float* __restrict__ bO,
                                       float* __restrict__ out, int H, int W, int total)
{
  int i = blockIdx.x * blockDim.x + threadIdx.x;
  if (i >= total) return;
  int hw = i % (H * W);
  int b  = i / (H * W);
  float a = bO[0];
  #pragma unroll 8
  for (int c = 0; c < 32; ++c)
    a += w[c] * top[((size_t)b * 32 + c) * H * W + hw];
  out[i] = 1.0f / (1.0f + __expf(-a));
}

// ======================================================================
// Host side
// ======================================================================
struct BlockP {
  const float *w1,*b1,*s1,*o1,*m1,*v1,*w2,*b2,*s2,*o2,*m2,*v2;
};
struct Net {
  const float* x;
  BlockP en1,en2,en3,en4,de4,de3,de2,de1,fin;
  const float *wd,*wu,*wr,*wl,*w_out,*b_out;
};

extern "C" void kernel_launch(void* const* d_in, const int* in_sizes, int n_in,
                              void* d_out, int out_size, void* d_ws, size_t ws_size,
                              hipStream_t stream) {
  (void)n_in; (void)out_size; (void)ws_size;
  auto F = [&](int i) { return (const float*)d_in[i]; };

  Net net;
  int i = 0;
  if (in_sizes[0] == 8 * 3 * 128 * 256) {
    // insertion-order flattening: x, then blocks {w1,b1,s1,o1,m1,v1,w2,b2,s2,o2,m2,v2}
    net.x = F(i++);
    auto blk = [&](BlockP& p) {
      p.w1=F(i++); p.b1=F(i++); p.s1=F(i++); p.o1=F(i++); p.m1=F(i++); p.v1=F(i++);
      p.w2=F(i++); p.b2=F(i++); p.s2=F(i++); p.o2=F(i++); p.m2=F(i++); p.v2=F(i++);
    };
    blk(net.en1); blk(net.en2); blk(net.en3); blk(net.en4);
    blk(net.de4); blk(net.de3); blk(net.de2); blk(net.de1); blk(net.fin);
    net.wd=F(i++); net.wu=F(i++); net.wr=F(i++); net.wl=F(i++);
    net.w_out=F(i++); net.b_out=F(i++);
  } else {
    // jax tree_leaves (sorted dict keys): params first, x last
    net.b_out = F(i++);
    auto blk = [&](BlockP& p) {
      p.b1=F(i++); p.b2=F(i++); p.m1=F(i++); p.m2=F(i++); p.o1=F(i++); p.o2=F(i++);
      p.s1=F(i++); p.s2=F(i++); p.v1=F(i++); p.v2=F(i++); p.w1=F(i++); p.w2=F(i++);
    };
    blk(net.de1); blk(net.de2); blk(net.de3); blk(net.de4);
    blk(net.en1); blk(net.en2); blk(net.en3); blk(net.en4); blk(net.fin);
    net.w_out=F(i++); net.wd=F(i++); net.wl=F(i++); net.wr=F(i++); net.wu=F(i++);
    net.x = F(i++);
  }

  // workspace arena (total ~197 MB)
  size_t off = 0;
  auto carve = [&](size_t bytes) {
    float* p = (float*)((char*)d_ws + off);
    off += (bytes + 255) & ~(size_t)255;
    return p;
  };
  float* EN1 = carve((size_t)8*32*128*256*4);
  float* EN2 = carve((size_t)8*64*64*128*4);
  float* EN3 = carve((size_t)8*128*32*64*4);
  float* EN4 = carve((size_t)8*256*16*32*4);
  float* T0  = carve((size_t)8*64*128*256*4);   // up2x outputs / pooled inputs (max 67 MB)
  float* T1  = carve((size_t)8*32*128*256*4);   // block outputs (max 33.5 MB)
  float* T2  = carve((size_t)8*32*128*256*4);   // intra-block mid buffer (max 33.5 MB)

  auto conv = [&](const float* i1, int C1, const float* i2, int C2, int Cout,
                  int H, int W, const float* wg, const float* bi,
                  const float* s, const float* o, const float* m, const float* v,
                  float* outp) {
    if ((H * W) % 256 == 0) {
      dim3 g(H * W / 256, Cout / BM, 8);
      conv3x3_bn_lrelu<4><<<g, 256, 0, stream>>>(i1, C1, i2, C2, wg, bi, s, o, m, v,
                                                 outp, Cout, H, W);
    } else {
      dim3 g(H * W / 128, Cout / BM, 8);
      conv3x3_bn_lrelu<2><<<g, 256, 0, stream>>>(i1, C1, i2, C2, wg, bi, s, o, m, v,
                                                 outp, Cout, H, W);
    }
  };
  auto block = [&](const float* i1, int C1, const float* i2, int C2,
                   const BlockP& p, int Cmid, int H, int W, float* outp) {
    conv(i1, C1, i2, C2, Cmid, H, W, p.w1, p.b1, p.s1, p.o1, p.m1, p.v1, T2);
    conv(T2, Cmid, nullptr, 0, Cmid, H, W, p.w2, p.b2, p.s2, p.o2, p.m2, p.v2, outp);
  };
  auto pool = [&](const float* inp, float* outp, int C, int H, int W) {
    int total = 8 * C * (H / 2) * (W / 2);
    maxpool2_kernel<<<CDIV(total, 256), 256, 0, stream>>>(inp, outp, H, W, total);
  };
  auto up = [&](const float* inp, float* outp, int C, int H, int W) {
    int total = 8 * C * 2 * H * 2 * W;
    up2x_kernel<<<CDIV(total, 256), 256, 0, stream>>>(inp, outp, H, W, total);
  };

  // ---------------- encoder ----------------
  block(net.x, 3, nullptr, 0, net.en1, 32, 128, 256, EN1);
  pool(EN1, T0, 32, 128, 256);
  block(T0, 32, nullptr, 0, net.en2, 64, 64, 128, EN2);
  pool(EN2, T0, 64, 64, 128);
  block(T0, 64, nullptr, 0, net.en3, 128, 32, 64, EN3);
  pool(EN3, T0, 128, 32, 64);
  block(T0, 128, nullptr, 0, net.en4, 256, 16, 32, EN4);
  pool(EN4, T0, 256, 16, 32);
  // ---------------- decoder ----------------
  block(T0, 256, nullptr, 0, net.de4, 256, 8, 16, T1);
  up(T1, T0, 256, 8, 16);                                   // de4: 256 @ 16x32
  block(T0, 256, EN4, 256, net.de3, 256, 16, 32, T1);
  up(T1, T0, 256, 16, 32);                                  // de3: 256 @ 32x64
  block(T0, 256, EN3, 128, net.de2, 128, 32, 64, T1);
  up(T1, T0, 128, 32, 64);                                  // de2: 128 @ 64x128
  block(T0, 128, EN2, 64, net.de1, 64, 64, 128, T1);
  up(T1, T0, 64, 64, 128);                                  // de1: 64 @ 128x256
  block(T0, 64, EN1, 32, net.fin, 32, 128, 256, T1);        // top: 32 @ 128x256
  // ---------------- SCNN scans (in-place on T1) ----------------
  scan_rows_kernel<<<8, 256, 0, stream>>>(T1, net.wd, 128, 256, +1);
  scan_rows_kernel<<<8, 256, 0, stream>>>(T1, net.wu, 128, 256, -1);
  scan_cols_kernel<<<8, 256, 0, stream>>>(T1, net.wr, 128, 256, +1);
  scan_cols_kernel<<<8, 256, 0, stream>>>(T1, net.wl, 128, 256, -1);
  // ---------------- 1x1 conv + sigmoid ----------------
  int total = 8 * 128 * 256;
  outconv_sigmoid_kernel<<<CDIV(total, 256), 256, 0, stream>>>(
      T1, net.w_out, net.b_out, (float*)d_out, 128, 256, total);
}
